// AttentionForONNX_36395552866589
// MI455X (gfx1250) — matrix operand order repaired and checked by reading, not compiled, not run on hardware
//
#include <hip/hip_runtime.h>

typedef __bf16 bf16;
typedef __attribute__((ext_vector_type(8)))  bf16  bf16x8;
typedef __attribute__((ext_vector_type(16))) bf16  v16bf;
typedef __attribute__((ext_vector_type(8)))  float v8f;

#define S_ 128
#define B_ 16
#define E_ 1024
#define N_ 16
#define H_ 64
#define L_ 2048
#define T_ 2176
#define M_ 2048   // S_*B_

static __device__ __forceinline__ v16bf combine16(bf16x8 lo, bf16x8 hi) {
    return __builtin_shufflevector(lo, hi, 0,1,2,3,4,5,6,7,8,9,10,11,12,13,14,15);
}

// A-matrix fragment (16x32 bf16), base = row-major [16+][ld] region in LDS.
static __device__ __forceinline__ v16bf load_a_frag(const bf16* base, int ld, int lane) {
    int r  = lane & 15;
    int ko = (lane >> 4) * 8;
    const bf16* p = base + r * ld + ko;
    bf16x8 lo = *(const bf16x8*)(p);
    bf16x8 hi = *(const bf16x8*)(p + 16);
    return combine16(lo, hi);
}

// B-matrix fragment (32x16 bf16) where storage is N-major: Bs[n][k], k contiguous.
static __device__ __forceinline__ v16bf load_b_frag(const bf16* base, int ld, int lane) {
    int n  = lane & 15;
    int ko = (lane >> 4) * 8;
    const bf16* p = base + n * ld + ko;
    bf16x8 lo = *(const bf16x8*)(p);
    bf16x8 hi = *(const bf16x8*)(p + 16);
    return combine16(lo, hi);
}

static __device__ __forceinline__ v8f wmma_bf16(v16bf a, v16bf b, v8f c) {
    return __builtin_amdgcn_wmma_f32_16x16x32_bf16(false, a, false, b, (short)0, c, false, false);
}

// ---------------------------------------------------------------------------
// CDNA5 async global->LDS copy (ASYNCcnt-tracked, bypasses VGPRs).
// VDST operand = wave-relative LDS byte address (= low 32 bits of flat addr).
// ---------------------------------------------------------------------------
static __device__ __forceinline__ void async_copy_b128(const void* lds_ptr, const void* gptr) {
    unsigned lds_off = (unsigned)(unsigned long long)lds_ptr;
    unsigned long long ga = (unsigned long long)gptr;
    asm volatile("global_load_async_to_lds_b128 %0, %1, off"
                 :: "v"(lds_off), "v"(ga) : "memory");
}
static __device__ __forceinline__ void wait_async0() {
    asm volatile("s_wait_asynccnt 0" ::: "memory");
}
static __device__ __forceinline__ void wait_async3() {
    asm volatile("s_wait_asynccnt 3" ::: "memory");
}

// ---------------------------------------------------------------------------
// DPP16 ROW_XMASK butterfly within each 16-lane row (no LDS, no dscnt waits).
// ---------------------------------------------------------------------------
template <int XMASK>
static __device__ __forceinline__ float dpp_row_xor(float v) {
    int xi = __float_as_int(v);
    int y  = __builtin_amdgcn_update_dpp(xi, xi, 0x160 + XMASK, 0xf, 0xf, true);
    return __int_as_float(y);
}
static __device__ __forceinline__ float row16_reduce_max(float v) {
    v = fmaxf(v, dpp_row_xor<1>(v));
    v = fmaxf(v, dpp_row_xor<2>(v));
    v = fmaxf(v, dpp_row_xor<4>(v));
    v = fmaxf(v, dpp_row_xor<8>(v));
    return v;
}
static __device__ __forceinline__ float row16_reduce_add(float v) {
    v += dpp_row_xor<1>(v);
    v += dpp_row_xor<2>(v);
    v += dpp_row_xor<4>(v);
    v += dpp_row_xor<8>(v);
    return v;
}

// ---------------------------------------------------------------------------
// f32 -> bf16 conversion (grid-stride)
// ---------------------------------------------------------------------------
__global__ __launch_bounds__(256) void cvt_bf16_kernel(const float* __restrict__ src,
                                                       bf16* __restrict__ dst, int n) {
    int i = blockIdx.x * blockDim.x + threadIdx.x;
    int stride = gridDim.x * blockDim.x;
    for (; i < n; i += stride) dst[i] = (bf16)src[i];
}

// ---------------------------------------------------------------------------
// Old-cache -> new-cache copy (pure bandwidth, float4)
// ---------------------------------------------------------------------------
__global__ __launch_bounds__(256) void copy_cache_kernel(const float* __restrict__ kc,
                                                         const float* __restrict__ vc,
                                                         float* __restrict__ kdst,
                                                         float* __restrict__ vdst) {
    const size_t LH = (size_t)L_ * H_;
    const size_t TH = (size_t)T_ * H_;
    const size_t total4 = (size_t)256 * LH / 4;
    size_t i = (size_t)blockIdx.x * blockDim.x + threadIdx.x;
    size_t stride = (size_t)gridDim.x * blockDim.x;
    for (; i < total4; i += stride) {
        size_t f   = i * 4;
        size_t seg = f / LH;
        size_t off = f - seg * LH;
        float4 a = ((const float4*)kc)[i];
        float4 b = ((const float4*)vc)[i];
        *(float4*)(kdst + seg * TH + off) = a;
        *(float4*)(vdst + seg * TH + off) = b;
    }
}

// ---------------------------------------------------------------------------
// GEMM: Y = A(bf16 [M_,E_]) @ W(bf16 [E_,E_])^T + bias
// Double-buffered async global->LDS staging; 3 async b128 copies per thread
// per tile; s_wait_asynccnt 3 drains the previous tile while next is in flight.
// ---------------------------------------------------------------------------
template <int MODE>
__global__ __launch_bounds__(256) void gemm_bias_kernel(const bf16* __restrict__ A,
                                                        const bf16* __restrict__ W,
                                                        const float* __restrict__ bias,
                                                        void* __restrict__ dst,
                                                        float scale) {
    __shared__ bf16 Atile[2][128][32];
    __shared__ bf16 Btile[2][64][32];

    const int tid  = threadIdx.x;
    const int lane = tid & 31;
    const int wave = tid >> 5;
    const int m0   = blockIdx.x * 128;
    const int n0   = blockIdx.y * 64;

    const int ar = tid >> 1, ac = (tid & 1) * 16;
    const int br = tid >> 2, bc = (tid & 3) * 8;
    const bf16* aSrc = A + (size_t)(m0 + ar) * E_ + ac;
    const bf16* bSrc = W + (size_t)(n0 + br) * E_ + bc;

    v8f acc[4] = {v8f{}, v8f{}, v8f{}, v8f{}};

    // issue tile 0
    async_copy_b128(&Atile[0][ar][ac],     aSrc);
    async_copy_b128(&Atile[0][ar][ac + 8], aSrc + 8);
    async_copy_b128(&Btile[0][br][bc],     bSrc);

    const int NT = E_ / 32;
    for (int kk = 0; kk < NT; ++kk) {
        const int buf = kk & 1;
        if (kk + 1 < NT) {   // prefetch next tile into other buffer
            const int k1 = (kk + 1) * 32;
            async_copy_b128(&Atile[buf ^ 1][ar][ac],     aSrc + k1);
            async_copy_b128(&Atile[buf ^ 1][ar][ac + 8], aSrc + k1 + 8);
            async_copy_b128(&Btile[buf ^ 1][br][bc],     bSrc + k1);
            wait_async3();   // previous tile's 3 copies done (in-order per wave)
        } else {
            wait_async0();
        }
        __syncthreads();     // publish tile kk to all waves

        v16bf afrag = load_a_frag(&Atile[buf][wave * 16][0], 32, lane);
#pragma unroll
        for (int j = 0; j < 4; ++j) {
            v16bf bfrag = load_b_frag(&Btile[buf][j * 16][0], 32, lane);
            acc[j] = wmma_bf16(afrag, bfrag, acc[j]);
        }
        __syncthreads();     // all reads of buf done before it is re-staged
    }

    // Epilogue. C layout: vgpr i, lane l -> row = (l<16 ? i : i+8), col = l&15
    const int rbase = m0 + wave * 16 + ((lane >> 4) * 8);
    const int cl    = lane & 15;
#pragma unroll
    for (int j = 0; j < 4; ++j) {
        const int col = n0 + j * 16 + cl;
        const float bv = bias[col];
#pragma unroll
        for (int i = 0; i < 8; ++i) {
            const int row = rbase + i;
            float val = acc[j][i] + bv;
            if (MODE == 0) {
                ((bf16*)dst)[(size_t)row * E_ + col] = (bf16)(val * scale);
            } else if (MODE == 1) {
                int s = row >> 4, b = row & 15;       // row = s*B_ + b
                int nh = col >> 6, h = col & 63;      // col = nh*H_ + h
                ((float*)dst)[((size_t)(b * N_ + nh) * T_ + (L_ + s)) * H_ + h] = val;
            } else {
                ((float*)dst)[(size_t)row * E_ + col] = val;
            }
        }
    }
}

// ---------------------------------------------------------------------------
// Flash-style attention. One block per (b, head). 8 waves; wave w owns q rows
// 16w..16w+15. T processed in chunks of 64. q pre-scaled by H^-0.5.
// LDS: Qt 16K + Kt 8K + Vt 8K + Pt 16K = 48 KB.
// ---------------------------------------------------------------------------
__global__ __launch_bounds__(256) void attention_kernel(const bf16* __restrict__ qbuf,  // [M_,E_]
                                                        const float* __restrict__ Kc,   // [256,T_,H_]
                                                        const float* __restrict__ Vc,   // [256,T_,H_]
                                                        bf16* __restrict__ attn_out) {  // [M_,E_]
    __shared__ bf16 Qt[128][64];
    __shared__ bf16 Kt[64][64];       // [t][h]
    __shared__ bf16 Vt[64][64];       // [h][t]  (transposed)
    __shared__ bf16 Pt[8][16][64];    // per-wave probs scratch

    const int bn   = blockIdx.x;          // b*N_ + head
    const int b    = bn >> 4;
    const int nh   = bn & 15;
    const int tid  = threadIdx.x;
    const int lane = tid & 31;
    const int wave = tid >> 5;

    {   // async-load Q tile: Qt[s][h] = qbuf[(s*B_+b)*E_ + nh*H_ + h]
        int s = tid >> 1;
        int c = (tid & 1) * 32;
        const bf16* src = qbuf + (size_t)(s * B_ + b) * E_ + nh * H_ + c;
#pragma unroll
        for (int u = 0; u < 4; ++u)
            async_copy_b128(&Qt[s][c + u * 8], src + u * 8);
        wait_async0();
    }
    __syncthreads();

    const v16bf qf0 = load_a_frag(&Qt[wave * 16][0],  64, lane);  // h 0..31
    const v16bf qf1 = load_a_frag(&Qt[wave * 16][32], 64, lane);  // h 32..63

    v8f o[4] = {v8f{}, v8f{}, v8f{}, v8f{}};
    float mrow[8], lrow[8];
#pragma unroll
    for (int i = 0; i < 8; ++i) { mrow[i] = -1e30f; lrow[i] = 0.0f; }

    const float* Kbase = Kc + (size_t)bn * T_ * H_;
    const float* Vbase = Vc + (size_t)bn * T_ * H_;

    for (int t0 = 0; t0 < T_; t0 += 64) {
        __syncthreads();
        {   // stage K chunk [64][64] f32 -> bf16 row-major; thread: row r, 16 cols
            int r = tid >> 2;
            int c = (tid & 3) * 16;
            const float4* src = (const float4*)(Kbase + (size_t)(t0 + r) * H_ + c);
#pragma unroll
            for (int q = 0; q < 2; ++q) {
                float4 a = src[2 * q], d = src[2 * q + 1];
                bf16x8 kb;
                kb[0]=(bf16)a.x; kb[1]=(bf16)a.y; kb[2]=(bf16)a.z; kb[3]=(bf16)a.w;
                kb[4]=(bf16)d.x; kb[5]=(bf16)d.y; kb[6]=(bf16)d.z; kb[7]=(bf16)d.w;
                *(bf16x8*)(&Kt[r][c + 8 * q]) = kb;
            }
        }
        {   // stage V chunk transposed: Vt[h][t]
            int r = tid >> 2;
            int c = (tid & 3) * 16;
            const float4* src = (const float4*)(Vbase + (size_t)(t0 + r) * H_ + c);
#pragma unroll
            for (int q = 0; q < 4; ++q) {
                float4 a = src[q];
                Vt[c + 4 * q + 0][r] = (bf16)a.x;
                Vt[c + 4 * q + 1][r] = (bf16)a.y;
                Vt[c + 4 * q + 2][r] = (bf16)a.z;
                Vt[c + 4 * q + 3][r] = (bf16)a.w;
            }
        }
        __syncthreads();

        // scores: four 16x16 tiles over t (K-dim = h = 64 -> 2 wmma each)
        v8f sc[4];
#pragma unroll
        for (int tt = 0; tt < 4; ++tt) {
            v8f s = {};
            s = wmma_bf16(qf0, load_b_frag(&Kt[tt * 16][0],  64, lane), s);
            s = wmma_bf16(qf1, load_b_frag(&Kt[tt * 16][32], 64, lane), s);
            sc[tt] = s;
        }

        // online softmax update (DPP row reductions, no LDS)
        float alpha[8];
#pragma unroll
        for (int i = 0; i < 8; ++i) {
            float mx = fmaxf(fmaxf(sc[0][i], sc[1][i]), fmaxf(sc[2][i], sc[3][i]));
            mx = row16_reduce_max(mx);
            float mnew = fmaxf(mrow[i], mx);
            float p0 = __expf(sc[0][i] - mnew);
            float p1 = __expf(sc[1][i] - mnew);
            float p2 = __expf(sc[2][i] - mnew);
            float p3 = __expf(sc[3][i] - mnew);
            sc[0][i] = p0; sc[1][i] = p1; sc[2][i] = p2; sc[3][i] = p3;
            float ps = row16_reduce_add((p0 + p1) + (p2 + p3));
            alpha[i] = __expf(mrow[i] - mnew);
            lrow[i]  = lrow[i] * alpha[i] + ps;
            mrow[i]  = mnew;
        }
#pragma unroll
        for (int j = 0; j < 4; ++j)
#pragma unroll
            for (int i = 0; i < 8; ++i) o[j][i] *= alpha[i];

        {   // write probs (C layout) to per-wave LDS scratch as bf16
            int rb = (lane >> 4) * 8;
            int c  = lane & 15;
#pragma unroll
            for (int i = 0; i < 8; ++i) {
#pragma unroll
                for (int tt = 0; tt < 4; ++tt)
                    Pt[wave][rb + i][c + tt * 16] = (bf16)sc[tt][i];
            }
        }
        // wave-private region: same-wave LDS ordering + compiler waits suffice
        v16bf pf0 = load_a_frag(&Pt[wave][0][0],  64, lane);  // t 0..31
        v16bf pf1 = load_a_frag(&Pt[wave][0][32], 64, lane);  // t 32..63
#pragma unroll
        for (int j = 0; j < 4; ++j) {
            o[j] = wmma_bf16(pf0, load_b_frag(&Vt[j * 16][0],  64, lane), o[j]);
            o[j] = wmma_bf16(pf1, load_b_frag(&Vt[j * 16][32], 64, lane), o[j]);
        }
    }

    // normalize + store: attn_out[(s*B_+b)*E_ + nh*H_ + h]
    const int rb = wave * 16 + (lane >> 4) * 8;
    const int cl = lane & 15;
#pragma unroll
    for (int j = 0; j < 4; ++j) {
        const int h = j * 16 + cl;
#pragma unroll
        for (int i = 0; i < 8; ++i) {
            const int s = rb + i;
            float val = o[j][i] / lrow[i];
            attn_out[(size_t)(s * B_ + b) * E_ + nh * H_ + h] = (bf16)val;
        }
    }
}

// ---------------------------------------------------------------------------
extern "C" void kernel_launch(void* const* d_in, const int* in_sizes, int n_in,
                              void* d_out, int out_size, void* d_ws, size_t ws_size,
                              hipStream_t stream) {
    const float* query   = (const float*)d_in[0];
    // d_in[1] ("key") is unused by the reference (it projects k/v from query)
    const float* k_cache = (const float*)d_in[2];
    const float* v_cache = (const float*)d_in[3];
    const float* Wq = (const float*)d_in[4];
    const float* bq = (const float*)d_in[5];
    const float* Wk = (const float*)d_in[6];
    const float* bk = (const float*)d_in[7];
    const float* Wv = (const float*)d_in[8];
    const float* bv = (const float*)d_in[9];
    const float* Wo = (const float*)d_in[10];
    const float* bo = (const float*)d_in[11];

    float* out  = (float*)d_out;                        // [S,B,E]
    float* kdst = out + (size_t)S_ * B_ * E_;           // [B,N,T,H]
    float* vdst = kdst + (size_t)B_ * N_ * T_ * H_;     // [B,N,T,H]

    bf16* ws    = (bf16*)d_ws;
    bf16* q_bf  = ws;                                   // M_*E_
    bf16* Wq_bf = q_bf  + (size_t)M_ * E_;              // E_*E_
    bf16* Wk_bf = Wq_bf + (size_t)E_ * E_;
    bf16* Wv_bf = Wk_bf + (size_t)E_ * E_;
    bf16* Wo_bf = Wv_bf + (size_t)E_ * E_;
    bf16* qproj = Wo_bf + (size_t)E_ * E_;              // M_*E_ (scaled q heads)
    bf16* attn  = qproj + (size_t)M_ * E_;              // M_*E_

    // 1) f32 -> bf16 conversions
    cvt_bf16_kernel<<<2048, 256, 0, stream>>>(query, q_bf, M_ * E_);
    cvt_bf16_kernel<<<1024, 256, 0, stream>>>(Wq, Wq_bf, E_ * E_);
    cvt_bf16_kernel<<<1024, 256, 0, stream>>>(Wk, Wk_bf, E_ * E_);
    cvt_bf16_kernel<<<1024, 256, 0, stream>>>(Wv, Wv_bf, E_ * E_);
    cvt_bf16_kernel<<<1024, 256, 0, stream>>>(Wo, Wo_bf, E_ * E_);

    // 2) projections (WMMA, async double-buffered staging)
    dim3 gg(M_ / 128, E_ / 64);
    const float scaling = 0.125f;   // H^-0.5 = 64^-0.5
    gemm_bias_kernel<0><<<gg, 256, 0, stream>>>(q_bf, Wq_bf, bq, (void*)qproj, scaling);
    gemm_bias_kernel<1><<<gg, 256, 0, stream>>>(q_bf, Wk_bf, bk, (void*)kdst, 1.0f);
    gemm_bias_kernel<1><<<gg, 256, 0, stream>>>(q_bf, Wv_bf, bv, (void*)vdst, 1.0f);

    // 3) cache concat (bandwidth-bound)
    copy_cache_kernel<<<8192, 256, 0, stream>>>(k_cache, v_cache, kdst, vdst);

    // 4) attention over full T (reads caches from d_out)
    attention_kernel<<<B_ * N_, 256, 0, stream>>>(qproj, kdst, vdst, attn);

    // 5) output projection (WMMA)
    gemm_bias_kernel<2><<<gg, 256, 0, stream>>>(attn, Wo_bf, bo, (void*)out, 1.0f);
}
